// MoeBlock_1889785610748
// MI455X (gfx1250) — compile-verified, matrix operand order checked by reading, not run on hardware
//
#include <hip/hip_runtime.h>
#include <hip/hip_bf16.h>

// Problem constants (match reference)
#define NE     8           // num experts
#define TOPK   2
#define DDIM   1024
#define FDIM   4096
#define T0     1024        // B*S tokens
#define NPAIR  (T0*TOPK)   // 2048 (token,expert) pairs
#define CAP    NPAIR       // worst-case pairs per expert
#define MTILES 64          // CAP/32 m-tiles per expert

typedef __attribute__((ext_vector_type(16))) __bf16 v16bf;
typedef __attribute__((ext_vector_type(2)))  __bf16 v2bf;
typedef __attribute__((ext_vector_type(2)))  float  v2f;
typedef __attribute__((ext_vector_type(8)))  float  v8f;
typedef __attribute__((ext_vector_type(4)))  unsigned int v4u;
typedef __attribute__((ext_vector_type(4)))  int v4i;
typedef __attribute__((ext_vector_type(8)))  int v8i;

union FragU { v16bf v; unsigned u[8]; };

// float2 -> packed bf16x2; lowers to a single v_cvt_pk_bf16_f32 (no v_perm).
__device__ __forceinline__ unsigned pk_bf16(float lo, float hi) {
  v2f f; f.x = lo; f.y = hi;
  union { v2bf h; unsigned u; } p;
  p.h = __builtin_convertvector(f, v2bf);
  return p.u;
}

// ---------------------------------------------------------------- TDM helper
// Issue TENSOR_LOAD_TO_LDS of a 64(cols) x 32(rows) fp32 tile starting at
// `gaddr`, row stride `row_stride` elements, into LDS byte offset `lds_off`.
__device__ __forceinline__ void tdm_load_tile(const float* gaddr,
                                              unsigned lds_off,
                                              int row_stride) {
  unsigned long long ga = (unsigned long long)(size_t)gaddr;
  // D# group 0: count=1 | lds_addr | global_addr(57b) | type=2
  v4u g0 = { 1u, lds_off, (unsigned)ga,
             (unsigned)((ga >> 32) & 0x01FFFFFFull) | 0x80000000u };
  // D# group 1: data_size=4B; tensor_dim0=row_stride, tensor_dim1=32768 (no
  // clipping: tile always in-bounds); tile 64x32; dim0_stride=row_stride.
  v8i g1 = { (int)0x20000,
             (int)((unsigned)(row_stride & 0xFFFF) << 16),
             (int)(((unsigned)row_stride >> 16) | 0x80000000u),  // dim1 lo16=32768
             (int)(64 << 16),                                    // tile_dim0=64
             (int)32,                                            // tile_dim1=32
             (int)row_stride,                                    // stride lo32
             0, 0 };
  v4i gz = { 0, 0, 0, 0 };
#if defined(__clang_major__) && (__clang_major__ >= 23)
  v8i gz8 = { 0, 0, 0, 0, 0, 0, 0, 0 };
  __builtin_amdgcn_tensor_load_to_lds(g0, g1, gz, gz, gz8, 0);
#else
  __builtin_amdgcn_tensor_load_to_lds(g0, g1, gz, gz, 0);
#endif
}

// ---------------------------------------------------------------- zero init
__global__ void zero_init(float* __restrict__ out, int* __restrict__ cnt) {
  int g = blockIdx.x * blockDim.x + threadIdx.x;
  float4* o4 = (float4*)out;
  if (g < (T0 * DDIM) / 4) o4[g] = make_float4(0.f, 0.f, 0.f, 0.f);
  if (g < 16) cnt[g] = 0;
}

// ---------------------------------------------------------------- router
__global__ void router_kernel(const float* __restrict__ x,
                              const float* __restrict__ gk,
                              int* __restrict__ cnt,
                              int* __restrict__ list,
                              float* __restrict__ wpair) {
  int t = blockIdx.x * blockDim.x + threadIdx.x;
  if (t >= T0) return;
  const float* xr = x + (size_t)t * DDIM;
  float acc[NE];
#pragma unroll
  for (int e = 0; e < NE; ++e) acc[e] = 0.f;
  for (int d = 0; d < DDIM; ++d) {
    float xv = xr[d];
#pragma unroll
    for (int e = 0; e < NE; ++e) acc[e] += xv * gk[d * NE + e];
  }
  int i0 = 0; float g0 = acc[0];
#pragma unroll
  for (int e = 1; e < NE; ++e) if (acc[e] > g0) { g0 = acc[e]; i0 = e; }
  int i1 = -1; float g1 = -3.4e38f;
#pragma unroll
  for (int e = 0; e < NE; ++e) if (e != i0 && acc[e] > g1) { g1 = acc[e]; i1 = e; }
  float eb  = __expf(g1 - g0);
  float inv = __builtin_amdgcn_rcpf(1.f + eb);
  wpair[2 * t + 0] = inv;
  wpair[2 * t + 1] = eb * inv;
  int p0 = atomicAdd(&cnt[i0], 1); list[i0 * CAP + p0] = 2 * t + 0;
  int p1 = atomicAdd(&cnt[i1], 1); list[i1 * CAP + p1] = 2 * t + 1;
}

// Fragment helpers -----------------------------------------------------------
// A-fragment (16x32 bf16) from fp32 row `xr` at K-offset kk.
// Lanes 0-15: K = {0..7, 16..23}; lanes 16-31: K = {8..15, 24..31}.
__device__ __forceinline__ void load_a_frag(FragU& a, const float* xr, int kk,
                                            int half) {
  const float4* p0 = (const float4*)(xr + kk + half * 8);
  const float4* p1 = (const float4*)(xr + kk + 16 + half * 8);
  float4 f0 = p0[0], f1 = p0[1], f2 = p1[0], f3 = p1[1];
  a.u[0] = pk_bf16(f0.x, f0.y); a.u[1] = pk_bf16(f0.z, f0.w);
  a.u[2] = pk_bf16(f1.x, f1.y); a.u[3] = pk_bf16(f1.z, f1.w);
  a.u[4] = pk_bf16(f2.x, f2.y); a.u[5] = pk_bf16(f2.z, f2.w);
  a.u[6] = pk_bf16(f3.x, f3.y); a.u[7] = pk_bf16(f3.z, f3.w);
}

// B-fragment from fp32 LDS tile [32 K][64 cols]; bf16 pack at read time.
// K-adjacent rows are 256B apart -> ds_load_2addr_stride64_b32 mergeable.
__device__ __forceinline__ void load_b_frag_f32(FragU& b, const float* tile,
                                                int half, int ncol) {
#pragma unroll
  for (int j = 0; j < 8; ++j) {
    int k = ((j < 4) ? 2 * j : 16 + 2 * (j - 4)) + half * 8;
    b.u[j] = pk_bf16(tile[k * 64 + ncol], tile[(k + 1) * 64 + ncol]);
  }
}

__device__ __forceinline__ float silu_gate(float g, float u) {
  float ex = __expf(-g);
  return g * __builtin_amdgcn_rcpf(1.f + ex) * u;
}

// ---------------------------------------------------------------- GEMM 1
// H[pair, f] = silu(x@w0[e]) * (x@w1[e]); double-buffered TDM weight staging.
__global__ __launch_bounds__(256) void moe_gemm1(
    const float* __restrict__ x, const float* __restrict__ w0g,
    const float* __restrict__ w1g, const int* __restrict__ cnt,
    const int* __restrict__ list, float* __restrict__ H) {
  int e  = blockIdx.x >> 6;
  int m0 = (blockIdx.x & (MTILES - 1)) * 32;
  int n0 = blockIdx.y * 64;
  int ce = cnt[e];
  if (m0 >= ce) return;

  __shared__ float tb0[2][32 * 64];  // w0 tiles (double buffered), fp32
  __shared__ float tb1[2][32 * 64];  // w1 tiles
  __shared__ int rows[32];

  int tid = threadIdx.x;
  const float* W0 = w0g + (size_t)e * DDIM * FDIM;
  const float* W1 = w1g + (size_t)e * DDIM * FDIM;

  if (tid < 32) {
    int s = m0 + tid;
    rows[tid] = (s < ce) ? list[e * CAP + s] : -1;
    // Prologue: DMA first K-slab into buffer 0 (wave 0 drives the TDM).
    tdm_load_tile(W0 + n0, (unsigned)(size_t)(void*)&tb0[0][0], FDIM);
    tdm_load_tile(W1 + n0, (unsigned)(size_t)(void*)&tb1[0][0], FDIM);
  }
  __syncthreads();

  int lane = tid & 31;
  int wv   = tid >> 5;
  int wm   = wv & 1;
  int wn   = wv >> 1;
  int half = lane >> 4;
  int l15  = lane & 15;
  int ncol = wn * 16 + l15;

  int  myrow = rows[wm * 16 + l15];
  const float* xr = x + (size_t)((myrow >= 0) ? (myrow >> 1) : 0) * DDIM;

  v8f acc0 = {}; v8f acc1 = {};

  int ib = 0;
  for (int kk = 0; kk < DDIM; kk += 32, ib ^= 1) {
    if (tid < 32) __builtin_amdgcn_s_wait_tensorcnt(0);
    __syncthreads();  // buffer ib ready; buffer ib^1 free (readers done)
    if (tid < 32 && (kk + 32) < DDIM) {
      size_t nxt = (size_t)(kk + 32) * FDIM + n0;
      tdm_load_tile(W0 + nxt, (unsigned)(size_t)(void*)&tb0[ib ^ 1][0], FDIM);
      tdm_load_tile(W1 + nxt, (unsigned)(size_t)(void*)&tb1[ib ^ 1][0], FDIM);
    }
    __builtin_prefetch(xr + ((kk + 64 < DDIM) ? kk + 64 : kk), 0, 1);

    FragU a;   load_a_frag(a, xr, kk, half);
    FragU b0f; load_b_frag_f32(b0f, &tb0[ib][0], half, ncol);
    FragU b1f; load_b_frag_f32(b1f, &tb1[ib][0], half, ncol);

    acc0 = __builtin_amdgcn_wmma_f32_16x16x32_bf16(false, a.v, false, b0f.v,
                                                   (short)0, acc0, false, false);
    acc1 = __builtin_amdgcn_wmma_f32_16x16x32_bf16(false, a.v, false, b1f.v,
                                                   (short)0, acc1, false, false);
  }

  int col = n0 + ncol;
#pragma unroll
  for (int r = 0; r < 8; ++r) {
    int mrow = wm * 16 + half * 8 + r;
    int prow = rows[mrow];
    if (prow >= 0)
      H[(size_t)prow * FDIM + col] = silu_gate(acc0[r], acc1[r]);
  }
}

// ---------------------------------------------------------------- GEMM 2
// out[token] += weight * (H[pair] @ wo[e]); double-buffered TDM staging.
__global__ __launch_bounds__(256) void moe_gemm2(
    const float* __restrict__ H, const float* __restrict__ wog,
    const int* __restrict__ cnt, const int* __restrict__ list,
    const float* __restrict__ wpair, float* __restrict__ out) {
  int e  = blockIdx.x >> 6;
  int m0 = (blockIdx.x & (MTILES - 1)) * 32;
  int n0 = blockIdx.y * 64;
  int ce = cnt[e];
  if (m0 >= ce) return;

  __shared__ float tbo[2][32 * 64];  // wo tiles (double buffered), fp32
  __shared__ int rows[32];

  int tid = threadIdx.x;
  const float* WO = wog + (size_t)e * FDIM * DDIM;

  if (tid < 32) {
    int s = m0 + tid;
    rows[tid] = (s < ce) ? list[e * CAP + s] : -1;
    tdm_load_tile(WO + n0, (unsigned)(size_t)(void*)&tbo[0][0], DDIM);
  }
  __syncthreads();

  int lane = tid & 31;
  int wv   = tid >> 5;
  int wm   = wv & 1;
  int wn   = wv >> 1;
  int half = lane >> 4;
  int l15  = lane & 15;
  int ncol = wn * 16 + l15;

  int  myrow = rows[wm * 16 + l15];
  const float* hr = H + (size_t)((myrow >= 0) ? myrow : 0) * FDIM;

  v8f acc = {};

  int ib = 0;
  for (int kk = 0; kk < FDIM; kk += 32, ib ^= 1) {
    if (tid < 32) __builtin_amdgcn_s_wait_tensorcnt(0);
    __syncthreads();
    if (tid < 32 && (kk + 32) < FDIM) {
      size_t nxt = (size_t)(kk + 32) * DDIM + n0;
      tdm_load_tile(WO + nxt, (unsigned)(size_t)(void*)&tbo[ib ^ 1][0], DDIM);
    }
    __builtin_prefetch(hr + ((kk + 64 < FDIM) ? kk + 64 : kk), 0, 1);

    FragU a; load_a_frag(a, hr, kk, half);
    FragU b; load_b_frag_f32(b, &tbo[ib][0], half, ncol);

    acc = __builtin_amdgcn_wmma_f32_16x16x32_bf16(false, a.v, false, b.v,
                                                  (short)0, acc, false, false);
  }

  int col = n0 + ncol;
#pragma unroll
  for (int r = 0; r < 8; ++r) {
    int mrow = wm * 16 + half * 8 + r;
    int prow = rows[mrow];
    if (prow >= 0) {
      float v = wpair[prow] * acc[r];
      atomicAdd(&out[(size_t)(prow >> 1) * DDIM + col], v);
    }
  }
}

// ---------------------------------------------------------------- launch
extern "C" void kernel_launch(void* const* d_in, const int* in_sizes, int n_in,
                              void* d_out, int out_size, void* d_ws, size_t ws_size,
                              hipStream_t stream) {
  (void)in_sizes; (void)n_in; (void)out_size; (void)ws_size;
  const float* x   = (const float*)d_in[0];  // [B,S,D]
  const float* gk  = (const float*)d_in[1];  // [D,E]
  const float* w0  = (const float*)d_in[2];  // [E,D,F]
  const float* w1  = (const float*)d_in[3];  // [E,D,F]
  const float* wo  = (const float*)d_in[4];  // [E,F,D]
  float* out = (float*)d_out;                // [B,S,D] fp32

  int*   cnt   = (int*)d_ws;
  int*   list  = cnt + 16;
  float* wpair = (float*)(list + NE * CAP);
  float* Hbuf  = wpair + NPAIR;              // 32 MB fp32 intermediate

  zero_init<<<(T0 * DDIM / 4 + 255) / 256, 256, 0, stream>>>(out, cnt);
  router_kernel<<<(T0 + 255) / 256, 256, 0, stream>>>(x, gk, cnt, list, wpair);
  moe_gemm1<<<dim3(NE * MTILES, FDIM / 64), 256, 0, stream>>>(x, w0, w1, cnt, list, Hbuf);
  moe_gemm2<<<dim3(NE * MTILES, DDIM / 64), 256, 0, stream>>>(Hbuf, wo, cnt, list, wpair, out);
}
